// Net_22978075033873
// MI455X (gfx1250) — compile-verified, matrix operand order checked by reading, not run on hardware
//
#include <hip/hip_runtime.h>
#include <stdint.h>

typedef _Float16 f16;
typedef _Float16 v16h __attribute__((ext_vector_type(16)));
typedef float    v8f  __attribute__((ext_vector_type(8)));
typedef uint32_t u32x4 __attribute__((ext_vector_type(4)));
typedef uint32_t u32x8 __attribute__((ext_vector_type(8)));

#define Bsz   256
#define Nsmp  16000
#define Cch   32
#define Ktap  64
#define NBLK  125      // 128-sample half-window blocks
#define Tfrm  124
#define ANSF  10
#define CA    320      // C * ANS
#define HID   50
#define OUTN  10

#define PADROW 16064   // 32 zero-pad + 16000 + 32 zero-pad (f16, per batch row)
// workspace layout (bytes)
#define WS_BLOCKSUM 0
#define WS_AUDIOH   4096000            // 256*125*32*4
#define WS_KERH     (4096000 + 8224768)  // + 256*16064*2

__device__ __forceinline__ v8f wmma_f16(v16h a, v16h b, v8f c) {
  // D = A(16x32 f16) x B(32x16 f16) + C(16x16 f32)
  return __builtin_amdgcn_wmma_f32_16x16x32_f16(false, a, false, b, (short)0, c,
                                                false, false);
}

// K-offset within a 32-wide k-step for operand element e (ISA 7.12.2, 16-bit A):
// lanes 0-15 hold K={0..7,16..23}, lanes 16-31 hold K={8..15,24..31}
#define KOFF(e, lg) ((((e) & 8) << 1) | ((lg) << 3) | ((e) & 7))

// ---------------------------------------------------------------------------
// TDM: 1-D tile (nelem f16 elements) from global -> LDS, descriptor per
// ISA cdna5 §8 (D# group0: count/lds_addr/global_addr/type=2;
// group1: data_size=2B, tensor_dim0=tile_dim0=nelem, dim1=tile_dim1=1).
// Workgroup-uniform values -> "s" constraints give the SGPR quad/oct groups
// the VIMAGE encoding expects. Tracked with TENSORcnt.
// ---------------------------------------------------------------------------
__device__ __forceinline__ void tdm_load_1d(uint32_t lds_off, uint64_t gaddr,
                                            uint32_t nelem) {
  u32x4 g0;
  g0[0] = 1u;                                        // count=1, user mode
  g0[1] = lds_off;                                   // lds_addr (bytes)
  g0[2] = (uint32_t)(gaddr & 0xFFFFFFFFu);           // global_addr[31:0]
  g0[3] = (uint32_t)((gaddr >> 32) & 0x1FFFFFFu) | (2u << 30); // [56:32]|type=2
  u32x8 g1;
  g1[0] = 1u << 16;                                  // data_size=1 (2 bytes)
  g1[1] = (nelem & 0xFFFFu) << 16;                   // tensor_dim0[15:0]
  g1[2] = ((nelem >> 16) & 0xFFFFu) | (1u << 16);    // dim0[31:16], tensor_dim1=1
  g1[3] = (nelem & 0xFFFFu) << 16;                   // tile_dim0 = nelem
  g1[4] = 1u;                                        // tile_dim1=1, tile_dim2=0
  g1[5] = nelem;                                     // tensor_dim0_stride lo
  g1[6] = 0u;                                        // stride0 hi / stride1 lo
  g1[7] = 0u;                                        // tensor_dim1_stride hi
  asm volatile("tensor_load_to_lds %0, %1" :: "s"(g0), "s"(g1) : "memory");
}

// ---------------------------------------------------------------------------
// Kernel 0: stage f16 copies in workspace. Audio gets 32-elem zero pads per
// row so SAME-conv boundary handling is baked in and TDM tiles are aligned.
// ---------------------------------------------------------------------------
__global__ __launch_bounds__(256) void cvt_kernel(const float* __restrict__ audio,
                                                  const float* __restrict__ ker,
                                                  f16* __restrict__ audio_h,
                                                  f16* __restrict__ ker_h) {
  const long total = (long)Bsz * PADROW;
  for (long i = (long)blockIdx.x * blockDim.x + threadIdx.x; i < total;
       i += (long)gridDim.x * blockDim.x) {
    long b = i / PADROW;
    int  p = (int)(i - b * PADROW);
    int  s = p - 32;
    float v = (s >= 0 && s < Nsmp) ? audio[b * Nsmp + s] : 0.f;
    audio_h[i] = (f16)v;
  }
  int g = blockIdx.x * blockDim.x + threadIdx.x;
  if (g < Cch * Ktap) ker_h[g] = (f16)ker[g];
}

// ---------------------------------------------------------------------------
// Kernel 1: gammatone filterbank as WMMA GEMM + relu + 128-sample block sums.
// One workgroup = (batch b, block j). Wave 0 TDM-loads the signal window and
// the f16 bank into LDS; 8 waves each compute one 16-sample tile.
// blocksum layout: [(b*125 + j)*32 + c]
// ---------------------------------------------------------------------------
__global__ __launch_bounds__(256) void cochlea_kernel(const f16* __restrict__ audio_h,
                                                      const f16* __restrict__ ker_h,
                                                      float* __restrict__ blocksum) {
  __shared__ __align__(16) f16   xs[224];          // window: element i at xs[1+i]
  __shared__ __align__(16) f16   kh[Cch * Ktap];   // f16 gammatone bank
  __shared__ float csum[Cch];

  const int wg  = blockIdx.x;
  const int b   = wg / NBLK;
  const int j   = wg - b * NBLK;
  const int tid = threadIdx.x;
  const int lane = tid & 31;
  const int wave = tid >> 5;   // sample sub-tile 0..7 (16 samples each)
  const int lg   = lane >> 4;
  const int col  = lane & 15;

  if (wave == 0) {
    // window covers padded idx [j*128+1, j*128+191]; copy 200 from j*128 (16B ok)
    uint64_t ga = (uint64_t)(uintptr_t)audio_h +
                  ((uint64_t)b * PADROW + (uint64_t)j * 128u) * 2u;
    tdm_load_1d((uint32_t)(uintptr_t)xs, ga, 200u);
    tdm_load_1d((uint32_t)(uintptr_t)kh, (uint64_t)(uintptr_t)ker_h,
                (uint32_t)(Cch * Ktap));
    __builtin_amdgcn_s_wait_tensorcnt(0);
  }
  if (tid < Cch) csum[tid] = 0.f;
  __syncthreads();

  v8f acc0 = {};   // channels 0..15
  v8f acc1 = {};   // channels 16..31
  #pragma unroll
  for (int kk = 0; kk < 2; ++kk) {   // K=64 -> two 32-wide steps
    v16h av0, av1, bv;
    #pragma unroll
    for (int e = 0; e < 16; ++e) {
      const int k = kk * 32 + KOFF(e, lg);
      bv[e]  = xs[1 + wave * 16 + col + k];        // B[k, n=col] = Xpad[s+k]
      av0[e] = kh[col * Ktap + k];                 // A[m=col,      k]
      av1[e] = kh[(16 + col) * Ktap + k];          // A[m=16+col,   k]
    }
    acc0 = wmma_f16(av0, bv, acc0);
    acc1 = wmma_f16(av1, bv, acc1);
  }

  // relu + reduce the 16 columns of each wave, accumulate per-channel in LDS
  #pragma unroll
  for (int r = 0; r < 8; ++r) {
    float v0 = fmaxf(acc0[r], 0.f);
    float v1 = fmaxf(acc1[r], 0.f);
    #pragma unroll
    for (int off = 8; off; off >>= 1) {
      v0 += __shfl_xor(v0, off, 16);
      v1 += __shfl_xor(v1, off, 16);
    }
    if (col == 0) {
      const int row = lg ? (8 + r) : r;            // D: lanes16-31 hold M=8+r
      atomicAdd(&csum[row], v0);
      atomicAdd(&csum[16 + row], v1);
    }
  }
  __syncthreads();
  if (tid < Cch) blocksum[(b * NBLK + j) * Cch + tid] = csum[tid];
}

// ---------------------------------------------------------------------------
// Kernel 2: LIF SNN scan. One workgroup = 16 batch rows, walks all 124 steps.
// 128 threads = 4 waves; wave w owns hidden tile [w*16, w*16+16).
// All matmuls via WMMA f16->f32; membranes fp32 in LDS; spikes exact in f16.
// ---------------------------------------------------------------------------
__global__ __launch_bounds__(128) void snn_kernel(const float* __restrict__ bs,
                                                  const float* __restrict__ Wb,
                                                  const float* __restrict__ Wic,
                                                  const float* __restrict__ Wac,
                                                  float* __restrict__ out) {
  __shared__ f16   wb [HID * CA];    // [50][320]
  __shared__ f16   wic[HID * 64];    // [50][64]  (k padded w/ zeros)
  __shared__ f16   wac[OUTN * 64];   // [10][64]  (k padded w/ zeros)
  __shared__ f16   spa[16 * CA];     // AN spikes  [16][320]
  __shared__ f16   spb[16 * 64];     // bushy spikes, k-pad zero
  __shared__ f16   spc[16 * 64];     // IC spikes,    k-pad zero
  __shared__ float memb [16 * 64];
  __shared__ float memic[16 * 64];
  __shared__ float memac[16 * 16];

  const int tid  = threadIdx.x;
  const int lane = tid & 31;
  const int wave = tid >> 5;
  const int lg   = lane >> 4;
  const int col  = lane & 15;
  const int b0   = blockIdx.x * 16;

  for (int i = tid; i < HID * CA; i += 128) wb[i] = (f16)Wb[i];
  for (int i = tid; i < HID * 64; i += 128) {
    int hh = i >> 6, k = i & 63;
    wic[i] = (f16)((k < HID) ? Wic[hh * HID + k] : 0.f);
  }
  for (int i = tid; i < OUTN * 64; i += 128) {
    int hh = i >> 6, k = i & 63;
    wac[i] = (f16)((k < HID) ? Wac[hh * HID + k] : 0.f);
  }
  for (int i = tid; i < 16 * 64; i += 128) {
    memb[i] = 0.f; memic[i] = 0.f;
    spb[i] = (f16)0.f; spc[i] = (f16)0.f;
  }
  for (int i = tid; i < 16 * 16; i += 128) memac[i] = 0.f;
  __syncthreads();

  const int h  = wave * 16 + col;                 // hidden unit this lane owns
  const int hB = (h < HID)  ? h : (HID - 1);      // clamped B-row (cols>=50 unused)
  const int oB = (h < OUTN) ? h : (OUTN - 1);

  for (int t = 0; t < Tfrm; ++t) {
    // prefetch next frame's block sums (global_prefetch_b8)
    {
      int tn = (t + 2 <= NBLK - 1) ? (t + 2) : (NBLK - 1);
      __builtin_prefetch(&bs[((b0 + (tid & 15)) * NBLK + tn) * Cch], 0, 1);
    }
    // --- auditory nerve: env = (bs[t]+bs[t+1])/256, 10 scaled threshold fires
    for (int i = tid; i < 16 * CA; i += 128) {
      int bb = i / CA, jj = i - bb * CA;
      int c = jj / ANSF, a = jj - c * ANSF;
      float e0 = bs[((b0 + bb) * NBLK + t) * Cch + c];
      float e1 = bs[((b0 + bb) * NBLK + t + 1) * Cch + c];
      float x  = (e0 + e1) * (1.f / 256.f);
      float sc = 0.5f + (float)a * (1.f / 9.f);
      spa[i] = (f16)((x * sc - 0.5f) > 0.f ? 1.f : 0.f);
    }
    __syncthreads();

    // --- Bushy: cur[16x50] = spa[16x320] @ Wb^T, K=320 -> 10 k-steps
    v8f acc = {};
    for (int kk = 0; kk < 10; ++kk) {
      v16h av, bv;
      #pragma unroll
      for (int e = 0; e < 16; ++e) {
        const int k = kk * 32 + KOFF(e, lg);
        av[e] = spa[col * CA + k];
        bv[e] = wb[hB * CA + k];
      }
      acc = wmma_f16(av, bv, acc);
    }
    #pragma unroll
    for (int r = 0; r < 8; ++r) {
      int bb = lg ? (8 + r) : r;
      if (h < HID) {
        float m   = memb[bb * 64 + h] * 0.95f + acc[r];
        float spk = (m - 1.f) > 0.f ? 1.f : 0.f;
        m -= spk;                                   // subtract reset
        memb[bb * 64 + h] = m;
        spb[bb * 64 + h]  = (f16)spk;
      }
    }
    __syncthreads();

    // --- Inferior colliculus: K=50 (padded 64) -> 2 k-steps
    v8f acc2 = {};
    #pragma unroll
    for (int kk = 0; kk < 2; ++kk) {
      v16h av, bv;
      #pragma unroll
      for (int e = 0; e < 16; ++e) {
        const int k = kk * 32 + KOFF(e, lg);
        av[e] = spb[col * 64 + k];
        bv[e] = wic[hB * 64 + k];
      }
      acc2 = wmma_f16(av, bv, acc2);
    }
    #pragma unroll
    for (int r = 0; r < 8; ++r) {
      int bb = lg ? (8 + r) : r;
      if (h < HID) {
        float m   = memic[bb * 64 + h] * 0.95f + acc2[r];
        float spk = (m - 1.f) > 0.f ? 1.f : 0.f;
        m -= spk;
        memic[bb * 64 + h] = m;
        spc[bb * 64 + h]   = (f16)spk;
      }
    }
    __syncthreads();

    // --- Audio cortex: N=10 (one tile; wave 0 stores)
    v8f acc3 = {};
    #pragma unroll
    for (int kk = 0; kk < 2; ++kk) {
      v16h av, bv;
      #pragma unroll
      for (int e = 0; e < 16; ++e) {
        const int k = kk * 32 + KOFF(e, lg);
        av[e] = spc[col * 64 + k];
        bv[e] = wac[oB * 64 + k];
      }
      acc3 = wmma_f16(av, bv, acc3);
    }
    if (wave == 0 && col < OUTN) {
      #pragma unroll
      for (int r = 0; r < 8; ++r) {
        int bb    = lg ? (8 + r) : r;
        float m   = memac[bb * 16 + col] * 0.95f + acc3[r];
        float spk = (m - 1.f) > 0.f ? 1.f : 0.f;
        m -= spk;
        memac[bb * 16 + col] = m;
        size_t base = ((size_t)(b0 + bb) * Tfrm + t) * OUTN + col;
        out[base] = spk;                                   // spk_rec [B,T,OUT]
        out[(size_t)Bsz * Tfrm * OUTN + base] = m;         // mem_rec [B,T,OUT]
      }
    }
    __syncthreads();
  }
}

extern "C" void kernel_launch(void* const* d_in, const int* in_sizes, int n_in,
                              void* d_out, int out_size, void* d_ws, size_t ws_size,
                              hipStream_t stream) {
  (void)in_sizes; (void)n_in; (void)out_size; (void)ws_size;
  const float* audio = (const float*)d_in[0];   // [256,16000]
  const float* ker   = (const float*)d_in[1];   // [32,64]
  const float* Wb    = (const float*)d_in[2];   // [50,320]
  const float* Wic   = (const float*)d_in[3];   // [50,50]
  const float* Wac   = (const float*)d_in[4];   // [10,50]
  float* out = (float*)d_out;

  char* ws = (char*)d_ws;
  float* blocksum = (float*)(ws + WS_BLOCKSUM);  // 256*125*32 f32
  f16*   audio_h  = (f16*)(ws + WS_AUDIOH);      // [256][16064] zero-padded f16
  f16*   ker_h    = (f16*)(ws + WS_KERH);        // [32][64] f16

  cvt_kernel<<<2048, 256, 0, stream>>>(audio, ker, audio_h, ker_h);
  cochlea_kernel<<<Bsz * NBLK, 256, 0, stream>>>(audio_h, ker_h, blocksum);
  snn_kernel<<<Bsz / 16, 128, 0, stream>>>(blocksum, Wb, Wic, Wac, out);
}
